// MultiHeadAttention_16243566313954
// MI455X (gfx1250) — matrix operand order repaired and checked by reading, not compiled, not run on hardware
//
#include <hip/hip_runtime.h>
#include <hip/hip_bf16.h>

typedef __bf16 v16bf __attribute__((ext_vector_type(16)));
typedef __bf16 v2bf  __attribute__((ext_vector_type(2)));
typedef float  v8f   __attribute__((ext_vector_type(8)));
typedef int    v4i_t __attribute__((ext_vector_type(4)));

union Frag16 { v16bf v; unsigned int u[8]; };

#define EMB 1024
#define N3  3072
#define TSEQ 2048
#define NH  16
#define HD  64

#define HAS_ASYNC __has_builtin(__builtin_amdgcn_global_load_async_to_lds_b128)

// ---- bf16 helpers (hardware cvt via native __bf16 casts) -------------------
__device__ __forceinline__ unsigned short bf16s(float f) {
  __bf16 h = (__bf16)f;
  return __builtin_bit_cast(unsigned short, h);
}
__device__ __forceinline__ unsigned int pk2bf(float a, float b) {
  v2bf p; p.x = (__bf16)a; p.y = (__bf16)b;
  return __builtin_bit_cast(unsigned int, p);
}

// ---- async 16B global->LDS copy (CDNA5 ASYNCcnt path) ----------------------
__device__ __forceinline__ void copy16_async(const unsigned short* g,
                                             unsigned short* l) {
#if HAS_ASYNC
  __builtin_amdgcn_global_load_async_to_lds_b128(
      (__attribute__((address_space(1))) v4i_t*)g,
      (__attribute__((address_space(3))) v4i_t*)l, 0, 0);
#else
  *(uint4*)(void*)l = *(const uint4*)(const void*)g;
#endif
}
__device__ __forceinline__ void async_wait() {
#if HAS_ASYNC
#if __has_builtin(__builtin_amdgcn_s_wait_asynccnt)
  __builtin_amdgcn_s_wait_asynccnt(0);
#else
  asm volatile("s_wait_asynccnt 0" ::: "memory");
#endif
#endif
}

__device__ __forceinline__ v8f wmma_bf16(const Frag16& a, const Frag16& b, v8f c) {
  return __builtin_amdgcn_wmma_f32_16x16x32_bf16(false, a.v, false, b.v,
                                                 (short)0, c, false, false);
}

// ---------------------------------------------------------------------------
// Pre-pass 1: x f32 -> bf16 (pure copy thereafter)
// ---------------------------------------------------------------------------
__global__ __launch_bounds__(256) void convert_x_kernel(
    const float* __restrict__ X, unsigned short* __restrict__ Xb)
{
  const size_t g = ((size_t)blockIdx.x * 256 + threadIdx.x) * 8;
  float4 f0 = *(const float4*)(X + g);
  float4 f1 = *(const float4*)(X + g + 4);
  uint4 o;
  o.x = pk2bf(f0.x, f0.y); o.y = pk2bf(f0.z, f0.w);
  o.z = pk2bf(f1.x, f1.y); o.w = pk2bf(f1.z, f1.w);
  *(uint4*)(Xb + g) = o;
}

// ---------------------------------------------------------------------------
// Pre-pass 2: W [1024][3072] f32 -> Wt [3072][1024] bf16 (LDS tile transpose)
// ---------------------------------------------------------------------------
__global__ __launch_bounds__(256) void transpose_w_kernel(
    const float* __restrict__ W, unsigned short* __restrict__ Wt)
{
  __shared__ __align__(16) unsigned short tile[64][66];
  const int tid   = threadIdx.x;
  const int nbase = blockIdx.x * 64;
  const int kbase = blockIdx.y * 64;
  {
    const int kr = tid >> 2;
    const int n0 = (tid & 3) * 16;
    const float* src = W + (size_t)(kbase + kr) * N3 + nbase + n0;
#pragma unroll
    for (int q = 0; q < 4; ++q) {
      float4 f = *(const float4*)(src + q * 4);
      *(unsigned int*)&tile[kr][n0 + q * 4]     = pk2bf(f.x, f.y);
      *(unsigned int*)&tile[kr][n0 + q * 4 + 2] = pk2bf(f.z, f.w);
    }
  }
  __syncthreads();
  {
    const int n  = tid >> 2;
    const int k0 = (tid & 3) * 16;
    unsigned int o[8];
#pragma unroll
    for (int j = 0; j < 8; ++j)
      o[j] = (unsigned int)tile[k0 + 2 * j][n] |
             ((unsigned int)tile[k0 + 2 * j + 1][n] << 16);
    unsigned short* dst = Wt + (size_t)(nbase + n) * EMB + kbase + k0;
    *(uint4*)(dst)     = *(uint4*)&o[0];
    *(uint4*)(dst + 8) = *(uint4*)&o[4];
  }
}

// ---------------------------------------------------------------------------
// Kernel 1: qkv = x @ W + b, bf16 output. A/B tiles staged via async copies.
// ---------------------------------------------------------------------------
__global__ __launch_bounds__(256) void qkv_gemm_kernel(
    const unsigned short* __restrict__ Xb, const unsigned short* __restrict__ Wt,
    const float* __restrict__ bias, unsigned short* __restrict__ QKV)
{
  __shared__ __align__(16) unsigned short As[128][32]; // [row][k]
  __shared__ __align__(16) unsigned short Bs[128][32]; // [n][k]

  const int tid  = threadIdx.x;
  const int lane = tid & 31;
  const int wave = tid >> 5;
  const int ln15 = lane & 15;
  const int hi   = lane >> 4;
  const int mbase = blockIdx.y * 128;
  const int nbase = blockIdx.x * 128;
  const int wrow = (wave & 3) * 32;
  const int wcol = (wave >> 2) * 64;

  v8f c[2][4];
#pragma unroll
  for (int i = 0; i < 2; ++i)
#pragma unroll
    for (int j = 0; j < 4; ++j) c[i][j] = {};

  const int r  = tid >> 1;            // staging row 0..127
  const int c0 = (tid & 1) * 16;      // staging col 0 or 16

  for (int k0 = 0; k0 < EMB; k0 += 32) {
    // stage A (x tile) and B (Wt tile): pure bf16 copies -> async to LDS
    {
      const unsigned short* asrc = Xb + (size_t)(mbase + r) * EMB + k0 + c0;
      copy16_async(asrc,     &As[r][c0]);
      copy16_async(asrc + 8, &As[r][c0 + 8]);
      const unsigned short* bsrc = Wt + (size_t)(nbase + r) * EMB + k0 + c0;
      copy16_async(bsrc,     &Bs[r][c0]);
      copy16_async(bsrc + 8, &Bs[r][c0 + 8]);
    }
    async_wait();
    __syncthreads();

    Frag16 af[2], bfm[4];
#pragma unroll
    for (int i = 0; i < 2; ++i) {
      const unsigned short* rp = &As[wrow + i * 16 + ln15][0];
      const int k8 = hi * 8;
#pragma unroll
      for (int v = 0; v < 8; ++v) {
        const int kl = ((v & 4) ? 16 : 0) + 2 * (v & 3) + k8;
        af[i].u[v] = *(const unsigned int*)(rp + kl);
      }
    }
#pragma unroll
    for (int j = 0; j < 4; ++j) {
      const unsigned short* np = &Bs[wcol + j * 16 + ln15][0];
#pragma unroll
      for (int v = 0; v < 8; ++v)
        bfm[j].u[v] = *(const unsigned int*)(np + ((v & 4) ? 16 : 0) + 2 * (v & 3) + hi * 8);
    }
#pragma unroll
    for (int i = 0; i < 2; ++i)
#pragma unroll
      for (int j = 0; j < 4; ++j)
        c[i][j] = wmma_bf16(af[i], bfm[j], c[i][j]);
    __syncthreads();
  }

#pragma unroll
  for (int i = 0; i < 2; ++i) {
#pragma unroll
    for (int j = 0; j < 4; ++j) {
      const int col = nbase + wcol + j * 16 + ln15;
      const float bv = bias[col];
      const int row0 = mbase + wrow + i * 16 + hi * 8;
#pragma unroll
      for (int rr = 0; rr < 8; ++rr)
        QKV[(size_t)(row0 + rr) * N3 + col] = bf16s(c[i][j][rr] + bv);
    }
  }
}

// ---------------------------------------------------------------------------
// Kernel 2: causal flash attention. 1 block per (128 queries, head, batch).
// K chunk staged async; V transposed via registers; P via LDS round trip.
// ---------------------------------------------------------------------------
__global__ __launch_bounds__(256) void flash_attn_kernel(
    const unsigned short* __restrict__ QKV, float* __restrict__ Out)
{
  __shared__ __align__(16) unsigned short Ks[32][64];      // [key][d]
  __shared__ __align__(16) unsigned short Vt[64][32];      // [d][key]
  __shared__ __align__(16) unsigned short Ps[8][16][32];   // per-wave P

  const int tid  = threadIdx.x;
  const int lane = tid & 31;
  const int wave = tid >> 5;
  const int ln15 = lane & 15;
  const int hi   = lane >> 4;
  const int qt = blockIdx.x;
  const int h  = blockIdx.y;
  const int b  = blockIdx.z;
  const int qbase = qt * 128;
  const int q0 = qbase + wave * 16;
  const size_t tok0 = (size_t)b * TSEQ;
  const int qcol = h * HD;
  const int kcol = EMB + h * HD;
  const int vcol = 2 * EMB + h * HD;

  Frag16 qa[2];
  {
    const unsigned short* rp = QKV + (tok0 + q0 + ln15) * N3 + qcol;
    const int k8 = hi * 8;
#pragma unroll
    for (int dc = 0; dc < 2; ++dc)
#pragma unroll
      for (int v = 0; v < 8; ++v) {
        const int kl = dc * 32 + ((v & 4) ? 16 : 0) + 2 * (v & 3) + k8;
        qa[dc].u[v] = *(const unsigned int*)(rp + kl);
      }
  }

  v8f o[4];
#pragma unroll
  for (int j = 0; j < 4; ++j) o[j] = {};
  float m[8], l[8];
#pragma unroll
  for (int rr = 0; rr < 8; ++rr) { m[rr] = -__builtin_inff(); l[rr] = 0.0f; }

  const int rowg0 = q0 + hi * 8;

  const int kend = qbase + 128;
  for (int kb = 0; kb < kend; kb += 32) {
    {
      const int kk = tid >> 3;
      const int d0 = (tid & 7) * 8;
      // K: pure copy -> async to LDS
      copy16_async(QKV + (tok0 + kb + kk) * N3 + kcol + d0, &Ks[kk][d0]);
      // V: transpose through registers
      const unsigned short* vsrc = QKV + (tok0 + kb + kk) * N3 + vcol + d0;
      unsigned short tmp[8];
      *(uint4*)tmp = *(const uint4*)vsrc;
#pragma unroll
      for (int j = 0; j < 8; ++j) Vt[d0 + j][kk] = tmp[j];
    }
    async_wait();
    __syncthreads();

    if (kb <= q0 + 15) {
      v8f s[2]; s[0] = {}; s[1] = {};
#pragma unroll
      for (int nt = 0; nt < 2; ++nt) {
#pragma unroll
        for (int dc = 0; dc < 2; ++dc) {
          Frag16 kf;
          const unsigned short* np = &Ks[nt * 16 + ln15][dc * 32];
#pragma unroll
          for (int v = 0; v < 8; ++v)
            kf.u[v] = *(const unsigned int*)(np + ((v & 4) ? 16 : 0) + 2 * (v & 3) + hi * 8);
          s[nt] = wmma_bf16(qa[dc], kf, s[nt]);
        }
      }
#pragma unroll
      for (int rr = 0; rr < 8; ++rr) {
        const int rowg = rowg0 + rr;
        float s0 = s[0][rr] * 0.125f;
        float s1 = s[1][rr] * 0.125f;
        if (kb + ln15 > rowg)      s0 = -__builtin_inff();
        if (kb + 16 + ln15 > rowg) s1 = -__builtin_inff();
        float mx = fmaxf(s0, s1);
#pragma unroll
        for (int off = 1; off < 16; off <<= 1)
          mx = fmaxf(mx, __shfl_xor(mx, off, 32));
        const float mn = fmaxf(m[rr], mx);
        const float corr = __expf(m[rr] - mn);
        m[rr] = mn;
        const float p0 = __expf(s0 - mn);
        const float p1 = __expf(s1 - mn);
        s[0][rr] = p0; s[1][rr] = p1;
        float rs = p0 + p1;
#pragma unroll
        for (int off = 1; off < 16; off <<= 1)
          rs += __shfl_xor(rs, off, 32);
        l[rr] = l[rr] * corr + rs;
#pragma unroll
        for (int j = 0; j < 4; ++j) o[j][rr] *= corr;
      }
#pragma unroll
      for (int nt = 0; nt < 2; ++nt)
#pragma unroll
        for (int rr = 0; rr < 8; ++rr)
          Ps[wave][rr + hi * 8][nt * 16 + ln15] = bf16s(s[nt][rr]);
      asm volatile("s_wait_dscnt 0" ::: "memory");
      Frag16 pa;
      {
        const unsigned short* pp = &Ps[wave][ln15][0];
        const int k8 = hi * 8;
#pragma unroll
        for (int v = 0; v < 8; ++v) {
          const int kl = ((v & 4) ? 16 : 0) + 2 * (v & 3) + k8;
          pa.u[v] = *(const unsigned int*)(pp + kl);
        }
      }
#pragma unroll
      for (int j = 0; j < 4; ++j) {
        Frag16 vf;
        const unsigned short* vp = &Vt[j * 16 + ln15][0];
#pragma unroll
        for (int v = 0; v < 8; ++v)
          vf.u[v] = *(const unsigned int*)(vp + ((v & 4) ? 16 : 0) + 2 * (v & 3) + hi * 8);
        o[j] = wmma_bf16(pa, vf, o[j]);
      }
    }
    __syncthreads();
  }

#pragma unroll
  for (int rr = 0; rr < 8; ++rr) {
    const float inv = 1.0f / l[rr];
    const int tok = rowg0 + rr;
    float* op = Out + (tok0 + tok) * (size_t)EMB + h * HD;
#pragma unroll
    for (int j = 0; j < 4; ++j)
      op[j * 16 + ln15] = o[j][rr] * inv;
  }
}

extern "C" void kernel_launch(void* const* d_in, const int* in_sizes, int n_in,
                              void* d_out, int out_size, void* d_ws, size_t ws_size,
                              hipStream_t stream) {
  const float* x    = (const float*)d_in[0];
  const float* W    = (const float*)d_in[1];
  const float* bias = (const float*)d_in[2];
  float* out = (float*)d_out;

  // workspace layout (bf16 elements): qkv | xbf | Wt
  unsigned short* qkv = (unsigned short*)d_ws;                  // 4096*3072
  unsigned short* xbf = qkv + (size_t)4096 * N3;                // 4096*1024
  unsigned short* Wt  = xbf + (size_t)4096 * EMB;               // 3072*1024

  convert_x_kernel<<<dim3((2 * TSEQ * EMB) / (256 * 8)), 256, 0, stream>>>(x, xbf);
  transpose_w_kernel<<<dim3(N3 / 64, EMB / 64), 256, 0, stream>>>(W, Wt);

  dim3 gGrid(N3 / 128, (2 * TSEQ) / 128);
  qkv_gemm_kernel<<<gGrid, 256, 0, stream>>>(xbf, Wt, bias, qkv);

  dim3 aGrid(TSEQ / 128, NH, 2);
  flash_attn_kernel<<<aGrid, 256, 0, stream>>>(qkv, out);
}
